// GemmaAttention_84404697301698
// MI455X (gfx1250) — matrix operand order repaired and checked by reading, not compiled, not run on hardware
//
#include <hip/hip_runtime.h>
#include <hip/hip_bf16.h>

// ---------------------------------------------------------------------------
// GQA attention forward for MI455X (gfx1250, wave32, WMMA).
// All matmuls use v_wmma_f32_16x16x32_bf16 (f32 accumulate).
// ---------------------------------------------------------------------------

typedef unsigned short u16;
typedef __attribute__((ext_vector_type(16))) __bf16 v16bf;
typedef __attribute__((ext_vector_type(8)))  float  v8f;

#define BATCH 2
#define SEQ   2048
#define HID   2048
#define NH    8
#define HD    256
#define ROWS  (BATCH * SEQ)      // 4096
#define QKVN  (NH * HD + 2 * HD) // 2560 (Q 0..2047 | K 2048..2303 | V 2304..2559)

union FragBF {
    uint4 u4[2];
    v16bf v;
};

__device__ __forceinline__ v8f v8f_zero() {
    v8f z = {0.f, 0.f, 0.f, 0.f, 0.f, 0.f, 0.f, 0.f};
    return z;
}

__device__ __forceinline__ u16 f32_to_bf16(float f) {
    unsigned int u = __float_as_uint(f);
    unsigned int r = u + 0x7FFFu + ((u >> 16) & 1u);   // round-to-nearest-even
    return (u16)(r >> 16);
}

__device__ __forceinline__ float bf16_to_f32(u16 h) {
    return __uint_as_float(((unsigned int)h) << 16);
}

__device__ __forceinline__ v8f wmma_bf16(const FragBF& a, const FragBF& b, v8f c) {
    // D = A(16x32) x B(32x16) + C, f32 accumulate
    return __builtin_amdgcn_wmma_f32_16x16x32_bf16(
        /*neg_a=*/false, a.v, /*neg_b=*/false, b.v,
        /*c_mod=*/(short)0, c, /*reuse_a=*/false, /*reuse_b=*/false);
}

// ---------------------------------------------------------------------------
// 1) f32 -> bf16 bulk convert (4 elems / thread)
// ---------------------------------------------------------------------------
__global__ void cvt_f32_bf16_kernel(const float* __restrict__ src,
                                    u16* __restrict__ dst, int n4) {
    int i = blockIdx.x * blockDim.x + threadIdx.x;
    if (i >= n4) return;
    float4 f = ((const float4*)src)[i];
    unsigned int lo = (unsigned int)f32_to_bf16(f.x) | ((unsigned int)f32_to_bf16(f.y) << 16);
    unsigned int hi = (unsigned int)f32_to_bf16(f.z) | ((unsigned int)f32_to_bf16(f.w) << 16);
    ((uint2*)dst)[i] = make_uint2(lo, hi);
}

// ---------------------------------------------------------------------------
// 2) pack Wq|Wk|Wv -> bf16 [HID, 2560]
// ---------------------------------------------------------------------------
__global__ void pack_qkv_kernel(const float* __restrict__ Wq,
                                const float* __restrict__ Wk,
                                const float* __restrict__ Wv,
                                u16* __restrict__ W) {
    int idx = blockIdx.x * blockDim.x + threadIdx.x;   // HID*QKVN
    if (idx >= HID * QKVN) return;
    int col = idx % QKVN, row = idx / QKVN;
    float v;
    if (col < 2048)      v = Wq[row * 2048 + col];
    else if (col < 2304) v = Wk[row * 256 + (col - 2048)];
    else                 v = Wv[row * 256 + (col - 2304)];
    W[idx] = f32_to_bf16(v);
}

// ---------------------------------------------------------------------------
// 3) bf16 GEMM: C[M,N] = A[M,K] * B[K,N].  WG = 256 threads (8 waves, 4x2).
//    Block tile 128x128, wave tile 32x64 (2x4 WMMA C-tiles), K-step 32.
// ---------------------------------------------------------------------------
template <bool OUT_BF16>
__global__ __launch_bounds__(256) void gemm_bf16_kernel(
    const u16* __restrict__ A, const u16* __restrict__ Bm, void* __restrict__ C,
    int M, int N, int K, int lda, int ldb, int ldc) {
    __shared__ u16 As[128 * 40];    // 128 rows x 32 K  (stride 40 -> 16B rows)
    __shared__ u16 Bs[32 * 136];    // 32 K x 128 N     (stride 136)

    const int tid  = threadIdx.x;
    const int wave = tid >> 5, lane = tid & 31;
    const int lm = lane & 15, hs = lane >> 4;   // C/A layout half select
    const int waveM = wave >> 1, waveN = wave & 1;
    const int m0 = blockIdx.y * 128;
    const int n0 = blockIdx.x * 128;

    v8f acc[2][4];
#pragma unroll
    for (int mt = 0; mt < 2; mt++)
#pragma unroll
        for (int nt = 0; nt < 4; nt++) acc[mt][nt] = v8f_zero();

    for (int k0 = 0; k0 < K; k0 += 32) {
        __syncthreads();
        {   // stage A: row = t/2, K-chunk = (t&1)*16 (2 x b128)
            int r = tid >> 1, cch = (tid & 1) << 4;
            const uint4* s4 = (const uint4*)(A + (size_t)(m0 + r) * lda + k0 + cch);
            uint4* d4 = (uint4*)&As[r * 40 + cch];
            d4[0] = s4[0]; d4[1] = s4[1];
        }
        {   // stage B: row = t/8 (32 rows), 16-col chunk = (t&7)*16 (2 x b128)
            int r = tid >> 3, cch = (tid & 7) << 4;
            const uint4* s4 = (const uint4*)(Bm + (size_t)(k0 + r) * ldb + n0 + cch);
            uint4* d4 = (uint4*)&Bs[r * 136 + cch];
            d4[0] = s4[0]; d4[1] = s4[1];
        }
        __syncthreads();

        if (k0 + 32 < K) {   // prefetch next K-step tiles into cache
            __builtin_prefetch(A + (size_t)(m0 + (tid >> 1)) * lda + k0 + 32, 0, 0);
            __builtin_prefetch(Bm + (size_t)(k0 + 32 + (tid >> 3)) * ldb + n0 + ((tid & 7) << 4), 0, 0);
        }

        FragBF af[2];
#pragma unroll
        for (int mt = 0; mt < 2; mt++) {
            // 16-bit A layout: lane = M + 16*Khalf; K chunks {hs*8..+7},{16+hs*8..+7}
            const int row = waveM * 32 + mt * 16 + lm;
            af[mt].u4[0] = *(const uint4*)&As[row * 40 + hs * 8];
            af[mt].u4[1] = *(const uint4*)&As[row * 40 + 16 + hs * 8];
        }
#pragma unroll
        for (int nt = 0; nt < 4; nt++) {
            FragBF bf;   // B layout: lane holds N = lm, K = {hs*8..+7},{16+hs*8..+7}
#pragma unroll
            for (int s = 0; s < 16; s++)
                ((u16*)&bf)[s] = Bs[((s & 7) + 8 * hs + ((s >> 3) << 4)) * 136 + waveN * 64 + nt * 16 + lm];
#pragma unroll
            for (int mt = 0; mt < 2; mt++)
                acc[mt][nt] = wmma_bf16(af[mt], bf, acc[mt][nt]);
        }
    }

#pragma unroll
    for (int mt = 0; mt < 2; mt++)
#pragma unroll
        for (int nt = 0; nt < 4; nt++)
#pragma unroll
            for (int i = 0; i < 8; i++) {
                // C layout: VGPR i, lane -> M = i + 8*hs, N = lm
                int gr = m0 + waveM * 32 + mt * 16 + i + 8 * hs;
                int gc = n0 + waveN * 64 + nt * 16 + lm;
                float v = acc[mt][nt][i];
                if (OUT_BF16)
                    ((u16*)C)[(size_t)gr * ldc + gc] = f32_to_bf16(v);
                else
                    ((float*)C)[(size_t)gr * ldc + gc] = v;
            }
}

// ---------------------------------------------------------------------------
// 4) RoPE in-place on Q (8 heads) and K (1 head) regions of qkv buffer.
// ---------------------------------------------------------------------------
__global__ void rope_kernel(u16* __restrict__ qkv) {
    int idx = blockIdx.x * blockDim.x + threadIdx.x;
    int pair = idx & 127;
    int t = idx >> 7;
    int head = t % 9;
    int row = t / 9;
    if (row >= ROWS) return;
    int pos = row & (SEQ - 1);
    int base = (head < NH) ? head * HD : 2048;   // K head at col 2048
    size_t off = (size_t)row * QKVN + base + pair;
    float x1 = bf16_to_f32(qkv[off]);
    float x2 = bf16_to_f32(qkv[off + 128]);
    float inv = __expf(-(float)pair * (9.210340371976184f / 128.0f)); // 10000^(-pair/128)
    float ang = (float)pos * inv;
    float s, c;
    __sincosf(ang, &s, &c);
    qkv[off]       = f32_to_bf16(x1 * c - x2 * s);
    qkv[off + 128] = f32_to_bf16(x2 * c + x1 * s);
}

// ---------------------------------------------------------------------------
// 5) Flash attention, S^T formulation.  grid = (SEQ/64, BATCH*NH), 128 thr.
//    Each wave is fully independent: 16 query rows x 256 out dims.
//    S^T(32x16) = K(32x256) x Q^T(256x16): K A-frags come straight from
//    global (native row-major layout), Q^T built once in LDS and its 8
//    B-fragments pinned in registers.  V B-frags also load straight from
//    global (lane = dim column).  Only P does an LDS layout round-trip.
//    No block-wide barriers; softmax reductions are in-lane + 1 shfl.
// ---------------------------------------------------------------------------
__global__ __launch_bounds__(128) void flash_attn_kernel(
    const u16* __restrict__ qkv, u16* __restrict__ out) {
    __shared__ u16 Qt[4 * 256 * 16];   // per-wave Q^T: [dim 0..255][query 0..15]
    __shared__ u16 Ps[4 * 16 * 40];    // per-wave P tile (16 queries x 32 keys)

    const int tid = threadIdx.x, wave = tid >> 5, lane = tid & 31;
    const int lm = lane & 15, hs = lane >> 4;
    const int b = blockIdx.y >> 3;
    const int h = blockIdx.y & 7;
    const int qb = blockIdx.x * 64;
    const int rowBase = b * SEQ;
    const int qg = qb + wave * 16 + lm;          // this lane's query position

    // Build Q^T once: lane handles query lm, dim half hs*128..+127.
    {
        const u16* qp = qkv + (size_t)(rowBase + qg) * QKVN + h * HD + hs * 128;
        u16* qt = &Qt[wave * 4096];
#pragma unroll
        for (int c = 0; c < 16; c++) {
            uint4 v4 = *(const uint4*)(qp + c * 8);
            const u16* e = (const u16*)&v4;
#pragma unroll
            for (int q = 0; q < 8; q++)
                qt[(hs * 128 + c * 8 + q) * 16 + lm] = e[q];
        }
    }
    asm volatile("s_wait_dscnt 0" ::: "memory");   // wave-local LDS RAW fence

    // Pin the 8 Q^T B-fragments (reused for every key block).
    FragBF qbf[8];
#pragma unroll
    for (int kk = 0; kk < 8; kk++) {
#pragma unroll
        for (int s = 0; s < 16; s++)
            ((u16*)&qbf[kk])[s] = Qt[wave * 4096 + (kk * 32 + (s & 7) + 8 * hs + ((s >> 3) << 4)) * 16 + lm];
    }

    v8f oacc[16];
#pragma unroll
    for (int i = 0; i < 16; i++) oacc[i] = v8f_zero();
    float m = -1e30f, l = 0.f;                 // per-lane stats for query lm

    const float cs = 0.0625f * 1.44269504f;    // 1/sqrt(256) * log2(e)
    const int numJ = qb / 32 + 2;              // causal: keys 0 .. qb+63

    for (int j = 0; j < numJ; j++) {
        const int jb = j * 32;

        // S^T(32 keys x 16 queries): K A-frags direct from global.
        v8f st0 = v8f_zero(), st1 = v8f_zero();
#pragma unroll
        for (int kk = 0; kk < 8; kk++) {
            FragBF ka0, ka1;
            const size_t r0 = (size_t)(rowBase + jb + lm) * QKVN + 2048 + kk * 32;
            const size_t r1 = (size_t)(rowBase + jb + 16 + lm) * QKVN + 2048 + kk * 32;
            ka0.u4[0] = *(const uint4*)(qkv + r0 + hs * 8);
            ka0.u4[1] = *(const uint4*)(qkv + r0 + 16 + hs * 8);
            ka1.u4[0] = *(const uint4*)(qkv + r1 + hs * 8);
            ka1.u4[1] = *(const uint4*)(qkv + r1 + 16 + hs * 8);
            st0 = wmma_bf16(ka0, qbf[kk], st0);
            st1 = wmma_bf16(ka1, qbf[kk], st1);
        }

        // Online softmax: lane owns query lm; 16 key-elements in-lane,
        // combine the two halves with one shfl_xor(16).
        float e0[8], e1[8];
        float mx = -1e30f;
#pragma unroll
        for (int i = 0; i < 8; i++) {
            const int k0g = jb + i + 8 * hs;        // key of st0 element
            const int k1g = jb + 16 + i + 8 * hs;   // key of st1 element
            e0[i] = (k0g <= qg) ? st0[i] * cs : -1e30f;
            e1[i] = (k1g <= qg) ? st1[i] * cs : -1e30f;
            mx = fmaxf(mx, fmaxf(e0[i], e1[i]));
        }
        mx = fmaxf(mx, __shfl_xor(mx, 16));
        float mnew  = fmaxf(m, mx);
        float alpha = exp2f(m - mnew);
        float psum = 0.f;
#pragma unroll
        for (int i = 0; i < 8; i++) {
            float p0 = exp2f(e0[i] - mnew);
            float p1 = exp2f(e1[i] - mnew);
            psum += p0 + p1;
            // P store in A layout: [query lm][key]
            Ps[wave * 640 + lm * 40 + i + 8 * hs]      = f32_to_bf16(p0);
            Ps[wave * 640 + lm * 40 + 16 + i + 8 * hs] = f32_to_bf16(p1);
        }
        psum += __shfl_xor(psum, 16);
        l = l * alpha + psum;
        m = mnew;

        // Rescale O: row r = i + 8*hs needs alpha of query r (lane r holds it).
#pragma unroll
        for (int i = 0; i < 8; i++) {
            float a = __shfl(alpha, i + 8 * hs);
#pragma unroll
            for (int nt = 0; nt < 16; nt++) oacc[nt][i] *= a;
        }

        asm volatile("s_wait_dscnt 0" ::: "memory");   // P visible wave-wide
        FragBF pf;   // reload P as A fragment (16 queries x 32 keys)
        {
            const u16* r = &Ps[wave * 640 + lm * 40 + hs * 8];
            pf.u4[0] = *(const uint4*)(r);
            pf.u4[1] = *(const uint4*)(r + 16);
        }

        // O(16x256) += P(16x32) x V(32x256): V B-frags direct from global,
        // lane holds dim lm, keys {hs*8..+7},{16+hs*8..+7}.
#pragma unroll
        for (int nt = 0; nt < 16; nt++) {
            FragBF vb;
#pragma unroll
            for (int s = 0; s < 16; s++)
                ((u16*)&vb)[s] = qkv[(size_t)(rowBase + jb + (s & 7) + 8 * hs + ((s >> 3) << 4)) * QKVN + 2304 + nt * 16 + lm];
            oacc[nt] = wmma_bf16(pf, vb, oacc[nt]);
        }

        if (j + 1 < numJ) {   // prefetch next key block's K and V rows
            const size_t nxt = (size_t)(rowBase + jb + 32 + lane) * QKVN;
            __builtin_prefetch(qkv + nxt + 2048, 0, 0);
            __builtin_prefetch(qkv + nxt + 2304, 0, 0);
        }
    }

    // Normalize and write bf16 attn output [ROWS, NH*HD].
    const int gmRow = rowBase + qb + wave * 16;
#pragma unroll
    for (int i = 0; i < 8; i++) {
        float lv = __shfl(l, i + 8 * hs);   // l of query (i + 8*hs)
        float inv = 1.0f / lv;
#pragma unroll
        for (int nt = 0; nt < 16; nt++) {
            float v = oacc[nt][i] * inv;
            out[(size_t)(gmRow + i + 8 * hs) * (NH * HD) + h * HD + nt * 16 + lm] =
                f32_to_bf16(v);
        }
    }
}

// ---------------------------------------------------------------------------
// launch
// ---------------------------------------------------------------------------
extern "C" void kernel_launch(void* const* d_in, const int* in_sizes, int n_in,
                              void* d_out, int out_size, void* d_ws, size_t ws_size,
                              hipStream_t stream) {
    const float* x  = (const float*)d_in[0];
    const float* Wq = (const float*)d_in[1];
    const float* Wk = (const float*)d_in[2];
    const float* Wv = (const float*)d_in[3];
    const float* Wo = (const float*)d_in[4];
    // d_in[5] (additive causal mask) is implemented analytically in-kernel.
    float* out = (float*)d_out;

    u16* ws    = (u16*)d_ws;
    u16* xb    = ws;                                   // [4096,2048] bf16
    u16* wqkv  = xb    + (size_t)ROWS * HID;           // [2048,2560] bf16
    u16* wob   = wqkv  + (size_t)HID * QKVN;           // [2048,2048] bf16
    u16* qkv   = wob   + (size_t)HID * HID;            // [4096,2560] bf16
    u16* attnO = qkv   + (size_t)ROWS * QKVN;          // [4096,2048] bf16

    // 1) converts / pack
    cvt_f32_bf16_kernel<<<(ROWS * HID / 4 + 255) / 256, 256, 0, stream>>>(x, xb, ROWS * HID / 4);
    cvt_f32_bf16_kernel<<<(HID * HID / 4 + 255) / 256, 256, 0, stream>>>(Wo, wob, HID * HID / 4);
    pack_qkv_kernel<<<(HID * QKVN + 255) / 256, 256, 0, stream>>>(Wq, Wk, Wv, wqkv);

    // 2) QKV projection: [4096,2048] x [2048,2560] -> bf16
    gemm_bf16_kernel<true><<<dim3(QKVN / 128, ROWS / 128), 256, 0, stream>>>(
        xb, wqkv, qkv, ROWS, QKVN, HID, HID, QKVN, QKVN);

    // 3) RoPE on Q and K regions
    rope_kernel<<<(ROWS * 9 * 128) / 256, 256, 0, stream>>>(qkv);

    // 4) flash attention -> bf16 [4096, 2048]
    flash_attn_kernel<<<dim3(SEQ / 64, BATCH * NH), 128, 0, stream>>>(qkv, attnO);

    // 5) output projection: [4096,2048] x [2048,2048] -> f32 d_out
    gemm_bf16_kernel<false><<<dim3(HID / 128, ROWS / 128), 256, 0, stream>>>(
        attnO, wob, out, ROWS, HID, HID, HID, HID, HID);
}
